// MultiHead_53858889892340
// MI455X (gfx1250) — matrix operand order repaired and checked by reading, not compiled, not run on hardware
//
#include <hip/hip_runtime.h>
#include <hip/hip_bf16.h>

// ---------------------------------------------------------------------------
// MultiHead attention for MI455X (gfx1250, wave32, WMMA).
// DIM=1024, HEADS=16 (hd=64), SEQ=50, BATCH=512.
// GEMMs: v_wmma_f32_16x16x32_f16, 128x128 tiles, double-buffered async
// global->LDS staging (ASYNCcnt). Attention uses an LDS mask-bias matrix.
// ---------------------------------------------------------------------------

typedef _Float16 h4  __attribute__((ext_vector_type(4)));
typedef _Float16 h8  __attribute__((ext_vector_type(8)));
typedef _Float16 h16 __attribute__((ext_vector_type(16)));
typedef float    f8  __attribute__((ext_vector_type(8)));

#define DIM   1024
#define SEQ   50
#define BATCH 512
#define HEADS 16
#define HD    64
#define MROWS (BATCH * SEQ)                 // 25600
constexpr size_t ACT_ELEMS = (size_t)MROWS * DIM;   // 26,214,400 halves per activation
constexpr size_t W_ELEMS   = (size_t)DIM * DIM;     // 1,048,576 halves per weight

// ---- CDNA5 async global->LDS copy (ASYNCcnt-tracked), inline asm ----------
__device__ __forceinline__ void async_cp_b128(unsigned lds_off, const void* g) {
    asm volatile("global_load_async_to_lds_b128 %0, %1, off"
                 :: "v"(lds_off), "v"((unsigned long long)g) : "memory");
}
#define ASYNC_WAIT_4() asm volatile("s_wait_asynccnt 0x4" ::: "memory")
#define ASYNC_WAIT_0() asm volatile("s_wait_asynccnt 0x0" ::: "memory")

// ---- WMMA fragment loaders (match CDNA5 ISA 16-bit A/B VGPR layouts) -------
__device__ __forceinline__ h16 frag_a(const _Float16* base, int row0, int stride, int k0) {
    const int lane = threadIdx.x & 31;
    const int m = lane & 15, hb = lane >> 4;
    const _Float16* p = base + (size_t)(row0 + m) * stride + k0 + hb * 8;
    h8 lo = *(const h8*)(p);
    h8 hi = *(const h8*)(p + 16);
    h16 r;
#pragma unroll
    for (int i = 0; i < 8; ++i) { r[i] = lo[i]; r[i + 8] = hi[i]; }
    return r;
}

__device__ __forceinline__ h16 frag_b(const _Float16* baseT, int col0, int stride, int k0) {
    const int lane = threadIdx.x & 31;
    const int n = lane & 15, kh = lane >> 4;
    const _Float16* p = baseT + (size_t)(col0 + n) * stride + k0 + kh * 8;
    h8 lo = *(const h8*)(p);
    h8 hi = *(const h8*)(p + 16);
    h16 r;
#pragma unroll
    for (int i = 0; i < 8; ++i) { r[i] = lo[i]; r[i + 8] = hi[i]; }
    return r;
}

__device__ __forceinline__ f8 wmma_f16(h16 a, h16 b, f8 c) {
    return __builtin_amdgcn_wmma_f32_16x16x32_f16(false, a, false, b, (short)0, c, false, false);
}

// ---- Kernel 1: transpose + convert weights -> f16 W^T ([n][k] row-major) --
__global__ void __launch_bounds__(256) wtconv_kernel(
    const float* __restrict__ Wq, const float* __restrict__ Wk,
    const float* __restrict__ Wv, const float* __restrict__ Wo,
    _Float16* __restrict__ WhT)
{
    __shared__ float tile[32][33];
    const int which = blockIdx.z;
    const float* W = (which == 0) ? Wq : (which == 1) ? Wk : (which == 2) ? Wv : Wo;
    const int n0 = blockIdx.x * 32, k0 = blockIdx.y * 32;
    const int tx = threadIdx.x & 31, ty = threadIdx.x >> 5;
#pragma unroll
    for (int rr = 0; rr < 4; ++rr) {
        const int k = ty * 4 + rr;
        tile[k][tx] = W[(size_t)(k0 + k) * DIM + n0 + tx];
    }
    __syncthreads();
#pragma unroll
    for (int rr = 0; rr < 4; ++rr) {
        const int n = ty * 4 + rr;
        WhT[(size_t)which * W_ELEMS + (size_t)(n0 + n) * DIM + k0 + tx]
            = (_Float16)tile[tx][n];
    }
}

// ---- Kernel 2: fused LayerNorm (Q,K,V) -> f16, wave-shuffle reductions ----
__global__ void __launch_bounds__(256) ln_kernel(
    const float* __restrict__ Q, const float* __restrict__ K, const float* __restrict__ V,
    const float* __restrict__ g, const float* __restrict__ beta,
    _Float16* __restrict__ outBase)
{
    __shared__ float red[8];
    const int which = blockIdx.x / MROWS;
    const int row   = blockIdx.x % MROWS;
    const float* src = (which == 0) ? Q : (which == 1) ? K : V;
    const int t = threadIdx.x;
    const int lane = t & 31, wid = t >> 5;
    float4 x = ((const float4*)(src + (size_t)row * DIM))[t];

    float s = x.x + x.y + x.z + x.w;
#pragma unroll
    for (int off = 16; off > 0; off >>= 1) s += __shfl_xor(s, off, 32);
    if (lane == 0) red[wid] = s;
    __syncthreads();
    float tot = 0.0f;
#pragma unroll
    for (int i = 0; i < 8; ++i) tot += red[i];
    const float mean = tot * (1.0f / DIM);
    __syncthreads();

    const float dx = x.x - mean, dy = x.y - mean, dz = x.z - mean, dw = x.w - mean;
    float ss = dx * dx + dy * dy + dz * dz + dw * dw;
#pragma unroll
    for (int off = 16; off > 0; off >>= 1) ss += __shfl_xor(ss, off, 32);
    if (lane == 0) red[wid] = ss;
    __syncthreads();
    float vtot = 0.0f;
#pragma unroll
    for (int i = 0; i < 8; ++i) vtot += red[i];
    const float inv = rsqrtf(vtot * (1.0f / DIM) + 1e-5f);

    const int c = t * 4;
    _Float16* dst = outBase + (size_t)which * ACT_ELEMS + (size_t)row * DIM + c;
    h4 o;
    o[0] = (_Float16)(dx * inv * g[c + 0] + beta[c + 0]);
    o[1] = (_Float16)(dy * inv * g[c + 1] + beta[c + 1]);
    o[2] = (_Float16)(dz * inv * g[c + 2] + beta[c + 2]);
    o[3] = (_Float16)(dw * inv * g[c + 3] + beta[c + 3]);
    *(h4*)dst = o;
}

// ---- Kernel 3/5: WMMA GEMM, 128x128 tile, 8 waves (4M x 2N), K-step 32 ----
// Each wave owns a 32x64 strip: 2 A-frags x 4 B-frags = 8 WMMA per stage.
// Double-buffered async staging from A (row-major) and pre-transposed W^T.
// MODE 0: C_h = A_h * W^T_h + bias (three projections via blockIdx.z)
// MODE 1: out_f32 = A_h * W^T_h + bias + residual
template <int MODE>
__global__ void __launch_bounds__(256) gemm_kernel(
    const _Float16* __restrict__ Abase, const _Float16* __restrict__ WTbase,
    const float* __restrict__ bias0, const float* __restrict__ bias1,
    const float* __restrict__ bias2,
    _Float16* __restrict__ Oh, const float* __restrict__ res, float* __restrict__ Of)
{
    __shared__ alignas(16) _Float16 sA[2][128 * 40];   // [r][k], stride 40
    __shared__ alignas(16) _Float16 sB[2][128 * 40];   // [n][k], stride 40

    const int z  = blockIdx.z;
    const _Float16* A  = Abase  + (size_t)z * ACT_ELEMS;
    const _Float16* WT = WTbase + (size_t)z * W_ELEMS;
    const float* bias = (MODE == 1) ? bias0 : ((z == 0) ? bias0 : (z == 1) ? bias1 : bias2);

    const int bm = blockIdx.x, bn = blockIdx.y;
    const int tid = threadIdx.x;
    const int w = tid >> 5;
    const int waveM = w & 3, waveN = w >> 2;

    // staging roles: 512 16B-segments per operand tile, 2 per thread each
    const int r0 = tid >> 2,        s0 = tid & 3;
    const int r1 = (tid + 256) >> 2, s1 = (tid + 256) & 3;

    const _Float16* gA0 = A  + (size_t)(bm * 128 + r0) * DIM + s0 * 8;
    const _Float16* gA1 = A  + (size_t)(bm * 128 + r1) * DIM + s1 * 8;
    const _Float16* gB0 = WT + (size_t)(bn * 128 + r0) * DIM + s0 * 8;
    const _Float16* gB1 = WT + (size_t)(bn * 128 + r1) * DIM + s1 * 8;
    const unsigned lA0 = (unsigned)(unsigned long long)&sA[0][r0 * 40 + s0 * 8];
    const unsigned lA1 = (unsigned)(unsigned long long)&sA[0][r1 * 40 + s1 * 8];
    const unsigned lB0 = (unsigned)(unsigned long long)&sB[0][r0 * 40 + s0 * 8];
    const unsigned lB1 = (unsigned)(unsigned long long)&sB[0][r1 * 40 + s1 * 8];
    constexpr unsigned BUF_STRIDE = 128 * 40 * sizeof(_Float16);   // bytes

    f8 acc[2][4];
#pragma unroll
    for (int mi = 0; mi < 2; ++mi)
#pragma unroll
        for (int nt = 0; nt < 4; ++nt) acc[mi][nt] = (f8)(0.0f);

    // prologue: stage k0 = 0 into buffer 0 (4 async issues per wave)
    async_cp_b128(lA0, gA0);
    async_cp_b128(lA1, gA1);
    async_cp_b128(lB0, gB0);
    async_cp_b128(lB1, gB1);

    for (int i = 0; i < 32; ++i) {
        const int buf = i & 1;
        if (i + 1 < 32) {
            const int kn = (i + 1) * 32;
            const unsigned nb = (unsigned)(buf ^ 1) * BUF_STRIDE;
            async_cp_b128(lA0 + nb, gA0 + kn);
            async_cp_b128(lA1 + nb, gA1 + kn);
            async_cp_b128(lB0 + nb, gB0 + kn);
            async_cp_b128(lB1 + nb, gB1 + kn);
            ASYNC_WAIT_4();      // stage i drained (in-order), stage i+1 in flight
        } else {
            ASYNC_WAIT_0();
        }
        __syncthreads();         // all waves' stage-i data visible

        h16 a0 = frag_a(sA[buf], 32 * waveM,      40, 0);
        h16 a1 = frag_a(sA[buf], 32 * waveM + 16, 40, 0);
#pragma unroll
        for (int nt = 0; nt < 4; ++nt) {
            h16 b = frag_b(sB[buf], 64 * waveN + 16 * nt, 40, 0);
            acc[0][nt] = wmma_f16(a0, b, acc[0][nt]);
            acc[1][nt] = wmma_f16(a1, b, acc[1][nt]);
        }
        __syncthreads();         // stage-i buffer free before it is re-filled
    }

    // epilogue: D element r -> row (hb*8 + r), col = lane&15
    const int lane = tid & 31, n = lane & 15, hb = lane >> 4;
#pragma unroll
    for (int mi = 0; mi < 2; ++mi) {
#pragma unroll
        for (int nt = 0; nt < 4; ++nt) {
            const int gc = bn * 128 + 64 * waveN + 16 * nt + n;
            const float bv = bias[gc];
#pragma unroll
            for (int r = 0; r < 8; ++r) {
                const size_t gr = (size_t)bm * 128 + 32 * waveM + 16 * mi + hb * 8 + r;
                const float val = acc[mi][nt][r] + bv;
                if (MODE == 0) {
                    Oh[(size_t)z * ACT_ELEMS + gr * DIM + gc] = (_Float16)val;
                } else {
                    const size_t off = gr * DIM + gc;
                    Of[off] = val + res[off];
                }
            }
        }
    }
}

// ---- Kernel 4: per-(batch,head) attention, S padded 50->64, hd=64 ---------
__global__ void __launch_bounds__(128) attn_kernel(
    const _Float16* __restrict__ q, const _Float16* __restrict__ k,
    const _Float16* __restrict__ v, const unsigned char* __restrict__ mask,
    _Float16* __restrict__ ctx)
{
    __shared__ alignas(16) _Float16 sQ[64 * 72];    // [s][d]  row-major
    __shared__ alignas(16) _Float16 sKt[64 * 72];   // [j][d]  (k row-major)
    __shared__ alignas(16) _Float16 sVt[64 * 72];   // [d][s]  (v transposed)
    __shared__ float    sS[64 * 68];                // mask-bias then scores [i][j]
    __shared__ alignas(16) _Float16 sP[64 * 72];    // probs   [i][s]
    __shared__ alignas(4) unsigned char sM[SEQ * SEQ];  // compact mask (2500 B)

    const int b = blockIdx.x >> 4, h = blockIdx.x & 15;
    const int tid = threadIdx.x;

    // stage compact mask (2500 bytes = 625 dwords; b*2500 is 4B-aligned)
    {
        const unsigned* msrc = (const unsigned*)(mask + (size_t)b * (SEQ * SEQ));
        for (int idx = tid; idx < (SEQ * SEQ) / 4; idx += 128)
            ((unsigned*)sM)[idx] = msrc[idx];
    }
    // stage q / k^T / v^T (zero-pad rows 50..63)
    for (int idx = tid; idx < 512; idx += 128) {
        const int s = idx >> 3, d8 = idx & 7;
        h8 vq = (h8)((_Float16)0), vk = vq, vv = vq;
        if (s < SEQ) {
            const size_t off = ((size_t)(b * SEQ + s)) * DIM + h * HD + d8 * 8;
            vq = *(const h8*)(q + off);
            vk = *(const h8*)(k + off);
            vv = *(const h8*)(v + off);
        }
        *(h8*)(sQ + s * 72 + d8 * 8) = vq;
#pragma unroll
        for (int j = 0; j < 8; ++j) {
            sKt[s * 72 + d8 * 8 + j] = vk[j];
            sVt[(d8 * 8 + j) * 72 + s] = vv[j];
        }
    }
    __syncthreads();

    // expand mask into additive f32 bias (0 or -1e30), padded region -1e30,
    // fully branchless (clamped LDS address + cndmask)
    for (int t2 = tid; t2 < 4096; t2 += 128) {
        const int i = t2 >> 6, j = t2 & 63;
        const bool pad = (i >= SEQ) | (j >= SEQ);
        const int  a   = pad ? 0 : (i * SEQ + j);
        const bool msk = pad | (sM[a] != 0);
        sS[i * 68 + j] = msk ? -1e30f : 0.0f;
    }
    __syncthreads();

    const int w = tid >> 5;
    const int lane = tid & 31, n = lane & 15, hb = lane >> 4;
    const int i0 = 16 * w;

    // scores = q @ k^T  (K-dim = hd = 64, two 32-steps)
    f8 acc[4];
#pragma unroll
    for (int i = 0; i < 4; ++i) acc[i] = (f8)(0.0f);
#pragma unroll
    for (int ks = 0; ks < HD; ks += 32) {
        h16 a = frag_a(sQ, i0, 72, ks);
#pragma unroll
        for (int nt = 0; nt < 4; ++nt) {
            h16 bf = frag_b(sKt, 16 * nt, 72, ks);
            acc[nt] = wmma_f16(a, bf, acc[nt]);
        }
    }
    // scores = acc * 1/sqrt(64) + bias (unconditional LDS RMW)
#pragma unroll
    for (int nt = 0; nt < 4; ++nt) {
        const int j = 16 * nt + n;
#pragma unroll
        for (int r = 0; r < 8; ++r) {
            const int i = i0 + hb * 8 + r;
            sS[i * 68 + j] = fmaf(acc[nt][r], 0.125f, sS[i * 68 + j]);
        }
    }
    __syncthreads();

    // softmax per row (64 rows, thread t<64 owns row t)
    if (tid < 64) {
        const int row = tid;
        float m = -3.0e38f;
        for (int j = 0; j < 64; ++j) m = fmaxf(m, sS[row * 68 + j]);
        float sum = 0.0f;
        for (int j = 0; j < 64; ++j) {
            const float ev = __expf(sS[row * 68 + j] - m);
            sum += ev;
            sP[row * 72 + j] = (_Float16)ev;
        }
        const float inv = 1.0f / sum;
        for (int j = 0; j < 64; ++j)
            sP[row * 72 + j] = (_Float16)((float)sP[row * 72 + j] * inv);
    }
    __syncthreads();

    // ctx = P @ v  (K-dim = S padded 64, N = hd = 64)
    f8 acc2[4];
#pragma unroll
    for (int i = 0; i < 4; ++i) acc2[i] = (f8)(0.0f);
#pragma unroll
    for (int ks = 0; ks < 64; ks += 32) {
        h16 a = frag_a(sP, i0, 72, ks);
#pragma unroll
        for (int nt = 0; nt < 4; ++nt) {
            h16 bf = frag_b(sVt, 16 * nt, 72, ks);
            acc2[nt] = wmma_f16(a, bf, acc2[nt]);
        }
    }
#pragma unroll
    for (int nt = 0; nt < 4; ++nt) {
        const int d = 16 * nt + n;
#pragma unroll
        for (int r = 0; r < 8; ++r) {
            const int i = i0 + hb * 8 + r;
            if (i < SEQ)
                ctx[((size_t)(b * SEQ + i)) * DIM + h * HD + d] = (_Float16)acc2[nt][r];
        }
    }
}

// ---------------------------------------------------------------------------
extern "C" void kernel_launch(void* const* d_in, const int* in_sizes, int n_in,
                              void* d_out, int out_size, void* d_ws, size_t ws_size,
                              hipStream_t stream) {
    (void)in_sizes; (void)n_in; (void)out_size; (void)ws_size;

    const float* Q    = (const float*)d_in[0];
    const float* K    = (const float*)d_in[1];
    const float* V    = (const float*)d_in[2];
    const unsigned char* mask = (const unsigned char*)d_in[3];   // jnp.bool_ -> 1 byte
    const float* Wq   = (const float*)d_in[4];
    const float* bq   = (const float*)d_in[5];
    const float* Wk   = (const float*)d_in[6];
    const float* bk   = (const float*)d_in[7];
    const float* Wv   = (const float*)d_in[8];
    const float* bv   = (const float*)d_in[9];
    const float* Wo   = (const float*)d_in[10];
    const float* bo   = (const float*)d_in[11];
    const float* ln_g = (const float*)d_in[12];
    const float* ln_b = (const float*)d_in[13];
    float* out = (float*)d_out;

    _Float16* ws  = (_Float16*)d_ws;
    _Float16* WhT = ws;                            //  4 * 1M halves      ( 8 MB)
    _Float16* nkv = WhT + 4 * W_ELEMS;             //  3 * ACT halves     (157 MB)
    _Float16* prj = nkv + 3 * ACT_ELEMS;           //  3 * ACT halves     (157 MB)
    _Float16* ctx = nkv;                           //  reuse qn region after projections

    // 1) weights -> f16, transposed ([n][k] row-major)
    wtconv_kernel<<<dim3(32, 32, 4), 256, 0, stream>>>(Wq, Wk, Wv, Wo, WhT);
    // 2) LayerNorm Q,K,V -> f16
    ln_kernel<<<3 * MROWS, 256, 0, stream>>>(Q, K, V, ln_g, ln_b, nkv);
    // 3) three projection GEMMs (z = 0,1,2), async double-buffered
    gemm_kernel<0><<<dim3(MROWS / 128, DIM / 128, 3), 256, 0, stream>>>(
        nkv, WhT, bq, bk, bv, prj, nullptr, nullptr);
    // 4) attention per (batch, head)
    attn_kernel<<<BATCH * HEADS, 128, 0, stream>>>(
        prj, prj + ACT_ELEMS, prj + 2 * ACT_ELEMS, mask, ctx);
    // 5) output projection + bias + residual -> f32 out
    gemm_kernel<1><<<dim3(MROWS / 128, DIM / 128, 1), 256, 0, stream>>>(
        ctx, WhT + 3 * W_ELEMS, bo, bo, bo, nullptr, Q, out);
}
